// myBlock_26225070310146
// MI455X (gfx1250) — matrix operand order repaired and checked by reading, not compiled, not run on hardware
//
#include <hip/hip_runtime.h>
#include <hip/hip_bf16.h>
#include <math.h>

// ---------------------------------------------------------------------------
// MI455X (gfx1250) transformer block w/ token pruning.
// Compute-bound (~200 GFLOP vs ~70MB => AI ~3000 flop/B at 23.3 TB/s HBM):
// all GEMMs + attention on v_wmma_f32_16x16x32_bf16 (wave32), GEMM operands
// staged into LDS with GLOBAL_LOAD_ASYNC_TO_LDS_B128 (ASYNCcnt) double-buffer.
// ---------------------------------------------------------------------------

#define Bv   16
#define Nv   785
#define Cv   768
#define Hh   12
#define Dh   64
#define Mrows (Bv*Nv)          // 12560 = 785*16
#define Hdim (4*Cv)            // 3072
#define KEEP 589               // 1 CLS + 588 selected
#define SEL  588
#define LPAD 608               // KEEP padded to multiple of 32 (19 K-chunks)

typedef __attribute__((ext_vector_type(16))) __bf16        v16bf;
typedef __attribute__((ext_vector_type(8)))  float         v8f;
typedef __attribute__((ext_vector_type(8)))  unsigned int  v8u;

union Frag { v8u u; v16bf b; };

__device__ inline unsigned short f2bf(float f) {
  unsigned int x = __builtin_bit_cast(unsigned int, f);
  unsigned int r = (x + 0x7FFFu + ((x >> 16) & 1u)) >> 16;   // RNE
  return (unsigned short)r;
}
__device__ inline float bf2f(unsigned short u) {
  unsigned int x = ((unsigned int)u) << 16;
  return __builtin_bit_cast(float, x);
}

// A fragment (16x32 bf16). ISA 7.12.2: lane m=lane&15, half=lane>>4.
// pair p<4 -> K = half*8+2p+{0,1}; p>=4 -> K = 16+half*8+2(p-4)+{0,1}.
__device__ inline void load_a(Frag& f, const unsigned short* rowk0, int half) {
  const unsigned int* p32 = (const unsigned int*)rowk0;
  int b0 = half * 4;
  int b1 = 8 + half * 4;
#pragma unroll
  for (int p = 0; p < 4; p++) f.u[p]     = p32[b0 + p];
#pragma unroll
  for (int p = 0; p < 4; p++) f.u[4 + p] = p32[b1 + p];
}

// B fragment (32x16 bf16). lane n=lane&15; pair p<4 -> K=half*8+2p+{0,1}; p>=4 -> K=16+half*8+2(p-4)+{0,1}.
// B stored transposed (Bt[n][k], k contiguous) -> two runs of 4 dwords.
__device__ inline void load_b(Frag& f, const unsigned short* rowk0, int half) {
  const unsigned int* p32 = (const unsigned int*)(rowk0);
  int b0 = half * 4;
#pragma unroll
  for (int p = 0; p < 4; p++) f.u[p] = p32[b0 + p];
#pragma unroll
  for (int p = 0; p < 4; p++) f.u[4 + p] = p32[8 + b0 + p];
}

__device__ inline v8f wmma_bf16(const Frag& a, const Frag& b, v8f c) {
  return __builtin_amdgcn_wmma_f32_16x16x32_bf16(false, a.b, false, b.b,
                                                 (short)0, c, false, false);
}

// Low 32 bits of a generic pointer into __shared__ == wave-relative LDS byte
// offset (shared aperture: LDS address is the low 32 address bits).
__device__ inline unsigned lds_off32(const void* p) {
  return (unsigned)(unsigned long long)p;
}
// 16B global->LDS async copy (ASYNCcnt-tracked).
__device__ inline void async_cp16(const unsigned short* g, unsigned lds) {
  asm volatile("global_load_async_to_lds_b128 %0, %1, off"
               :: "v"(lds), "v"((unsigned long long)g)
               : "memory");
}
#if defined(__has_builtin)
#if __has_builtin(__builtin_amdgcn_s_wait_asynccnt)
#define WAIT_ASYNC(n) __builtin_amdgcn_s_wait_asynccnt(n)
#endif
#endif
#ifndef WAIT_ASYNC
#define WAIT_ASYNC(n) asm volatile("s_wait_asynccnt %0" :: "i"(n) : "memory")
#endif

// ---------------------------------------------------------------------------
// fp32 [K][N] weight -> bf16 transposed [N][K]
__global__ void cvt_transpose(const float* __restrict__ W,
                              unsigned short* __restrict__ Wt, int K, int N) {
  long long id = (long long)blockIdx.x * blockDim.x + threadIdx.x;
  long long total = (long long)K * N;
  if (id >= total) return;
  int n = (int)(id / K);
  int k = (int)(id % K);
  Wt[(long long)n * K + k] = f2bf(W[(long long)k * N + n]);
}

// ---------------------------------------------------------------------------
// LayerNorm over C=768, emit bf16
__global__ void layernorm_bf16(const float* __restrict__ x,
                               const float* __restrict__ g,
                               const float* __restrict__ be,
                               unsigned short* __restrict__ out) {
  const int row = blockIdx.x;
  const int tid = threadIdx.x;            // 256 threads, 3 cols each
  const float* xr = x + (long long)row * Cv;
  __shared__ float s1[256], s2[256];
  float v0 = xr[tid], v1 = xr[tid + 256], v2 = xr[tid + 512];
  s1[tid] = v0 + v1 + v2;
  s2[tid] = v0 * v0 + v1 * v1 + v2 * v2;
  __syncthreads();
  for (int off = 128; off > 0; off >>= 1) {
    if (tid < off) { s1[tid] += s1[tid + off]; s2[tid] += s2[tid + off]; }
    __syncthreads();
  }
  float mean = s1[0] * (1.f / Cv);
  float var  = s2[0] * (1.f / Cv) - mean * mean;
  float inv  = rsqrtf(var + 1e-5f);
  unsigned short* orow = out + (long long)row * Cv;
  int c = tid;        orow[c] = f2bf((v0 - mean) * inv * g[c] + be[c]);
  c = tid + 256;      orow[c] = f2bf((v1 - mean) * inv * g[c] + be[c]);
  c = tid + 512;      orow[c] = f2bf((v2 - mean) * inv * g[c] + be[c]);
}

// ---------------------------------------------------------------------------
// Macro-tiled WMMA GEMM: C[m][n] = sum_k A[m][k]*Bt[n][k] (+bias)(+res/GELU)
// Block 256 = 8 waves -> 64(M) x 128(N) macro tile; wave = 32x32 register
// block (4 accumulators). A/B staged in double-buffered LDS via
// global_load_async_to_lds_b128; 6 async b128 per thread per 64-wide K chunk.
// MODE 0: bf16 out. MODE 1: exact GELU -> bf16. MODE 2: res + acc -> f32.
template <int MODE>
__global__ void __launch_bounds__(256)
gemm_wmma(const unsigned short* __restrict__ A,
          const unsigned short* __restrict__ Bt,
          const float* __restrict__ bias,
          const float* __restrict__ res,
          unsigned short* __restrict__ outB,
          float* __restrict__ outF, int K, int N) {
  __shared__ __align__(16) unsigned short sA[2][64][64];   // 16 KB
  __shared__ __align__(16) unsigned short sB[2][128][64];  // 32 KB
  const int tid = threadIdx.x;
  const int lane = tid & 31;
  const int wave = tid >> 5;
  const int wm = wave & 1;        // 2 wave rows
  const int wn = wave >> 1;       // 4 wave cols
  const int half = lane >> 4, nl = lane & 15;
  const int mbase = blockIdx.x * 64;
  const int nbase = blockIdx.y * 128;
  const int nk = K / 64;

  auto stage = [&](int buf, int kc) {
    const int k0 = kc * 64;
#pragma unroll
    for (int j = 0; j < 2; j++) {          // A: 64 rows x 128B
      int ci = tid + j * 256;
      int row = ci >> 3, c = ci & 7;
      int rg = mbase + row;
      if (rg > Mrows - 1) rg = Mrows - 1;  // clamp; dead rows never stored
      async_cp16(A + (long long)rg * K + k0 + c * 8,
                 lds_off32(&sA[buf][row][c * 8]));
    }
#pragma unroll
    for (int j = 0; j < 4; j++) {          // B: 128 rows x 128B
      int ci = tid + j * 256;
      int row = ci >> 3, c = ci & 7;
      async_cp16(Bt + (long long)(nbase + row) * K + k0 + c * 8,
                 lds_off32(&sB[buf][row][c * 8]));
    }
  };

  stage(0, 0);
  if (nk > 1) stage(1, 1);

  v8f acc[2][2] = {};
  for (int kc = 0; kc < nk; kc++) {
    const int buf = kc & 1;
    if (kc == nk - 1) { WAIT_ASYNC(0); } else { WAIT_ASYNC(6); }
    __syncthreads();
#pragma unroll
    for (int ks = 0; ks < 2; ks++) {
      Frag a0, a1, b0, b1;
      load_a(a0, &sA[buf][wm * 32 + nl][ks * 32], half);
      load_a(a1, &sA[buf][wm * 32 + 16 + nl][ks * 32], half);
      load_b(b0, &sB[buf][wn * 32 + nl][ks * 32], half);
      load_b(b1, &sB[buf][wn * 32 + 16 + nl][ks * 32], half);
      acc[0][0] = wmma_bf16(a0, b0, acc[0][0]);
      acc[0][1] = wmma_bf16(a0, b1, acc[0][1]);
      acc[1][0] = wmma_bf16(a1, b0, acc[1][0]);
      acc[1][1] = wmma_bf16(a1, b1, acc[1][1]);
    }
    __syncthreads();                       // everyone done reading buf
    if (kc + 2 < nk) stage(buf, kc + 2);   // refill for chunk kc+2
  }

#pragma unroll
  for (int i = 0; i < 2; i++) {
#pragma unroll
    for (int j = 0; j < 2; j++) {
      const int col = nbase + wn * 32 + j * 16 + nl;
      const float bv = bias ? bias[col] : 0.f;
#pragma unroll
      for (int r = 0; r < 8; r++) {
        const int row = mbase + wm * 32 + i * 16 + r + 8 * half;
        if (row < Mrows) {
          float v = acc[i][j][r] + bv;
          if (MODE == 0) {
            outB[(long long)row * N + col] = f2bf(v);
          } else if (MODE == 1) {
            float gl = 0.5f * v * (1.0f + erff(v * 0.70710678118654752f));
            outB[(long long)row * N + col] = f2bf(gl);
          } else {
            outF[(long long)row * N + col] = res[(long long)row * N + col] + v;
          }
        }
      }
    }
  }
}

// ---------------------------------------------------------------------------
// scores[bh][t] = ||k_{t+1}|| / 64 for rem tokens t=0..783
__global__ void knorm_scores(const unsigned short* __restrict__ qkv,
                             float* __restrict__ scores) {
  int id = blockIdx.x * blockDim.x + threadIdx.x;
  if (id >= Bv * Hh * 784) return;
  int t = id % 784;
  int bh = id / 784;
  int h = bh % Hh, b = bh / Hh;
  const unsigned short* k =
      qkv + ((long long)(b * Nv + (t + 1)) * (3 * Cv) + Cv + h * Dh);
  float s = 0.f;
#pragma unroll 8
  for (int d = 0; d < Dh; d++) { float f = bf2f(k[d]); s += f * f; }
  scores[bh * 784 + t] = sqrtf(s) * (1.0f / Dh);
}

// ---------------------------------------------------------------------------
// Exact top-588 (jax tie-break: lower index first), output ascending indices.
__global__ void select_topk(const float* __restrict__ scores,
                            int* __restrict__ kidx) {
  const int bh = blockIdx.x;     // 192
  const int tid = threadIdx.x;   // 256
  __shared__ float ss[784];
  __shared__ unsigned char kp[784];
  for (int i = tid; i < 784; i += 256) ss[i] = scores[bh * 784 + i];
  __syncthreads();
  for (int i = tid; i < 784; i += 256) {
    float si = ss[i];
    int rank = 0;
    for (int j = 0; j < 784; j++) {
      float sj = ss[j];
      rank += (sj > si) || (sj == si && j < i);
    }
    kp[i] = (rank < SEL) ? 1 : 0;
  }
  __syncthreads();
  for (int i = tid; i < 784; i += 256) {
    if (kp[i]) {
      int pos = 0;
      for (int j = 0; j < i; j++) pos += kp[j];
      kidx[bh * SEL + pos] = i + 1;       // rem index -> token index
    }
  }
}

// ---------------------------------------------------------------------------
// Gather kept K,V: Kg[bh][slot][d] (bf16), VgT[bh][d][slot] (bf16), zero-pad.
__global__ void gather_kv(const unsigned short* __restrict__ qkv,
                          const int* __restrict__ kidx,
                          unsigned short* __restrict__ Kg,
                          unsigned short* __restrict__ VgT) {
  const int slot = blockIdx.x;  // 0..607
  const int bh = blockIdx.y;    // 192
  const int d = threadIdx.x;    // 64
  const int h = bh % Hh, b = bh / Hh;
  int tok = (slot == 0) ? 0 : (slot <= SEL ? kidx[bh * SEL + slot - 1] : -1);
  unsigned short kv = 0, vv = 0;
  if (tok >= 0) {
    long long base = (long long)(b * Nv + tok) * (3 * Cv) + h * Dh;
    kv = qkv[base + Cv + d];
    vv = qkv[base + 2 * Cv + d];
  }
  Kg[((long long)bh * LPAD + slot) * Dh + d] = kv;
  VgT[((long long)bh * Dh + d) * LPAD + slot] = vv;
}

// ---------------------------------------------------------------------------
// Fused attention: one wave per (b, h, 16-query-row tile).
// scores (16x608 f32) + probs (16x608 bf16) live in LDS (58KB << 320KB/WGP).
__global__ void __launch_bounds__(32)
attention_kernel(const unsigned short* __restrict__ qkv,
                 const unsigned short* __restrict__ Kg,
                 const unsigned short* __restrict__ VgT,
                 unsigned short* __restrict__ aout) {
  const int qt = blockIdx.x;    // 0..49 (785 rows, last tile clamped)
  const int h = blockIdx.y;
  const int b = blockIdx.z;
  const int bh = b * Hh + h;
  const int lane = threadIdx.x;
  const int half = lane >> 4;
  const int nl = lane & 15;
  __shared__ __align__(16) float sS[16][LPAD];
  __shared__ __align__(16) unsigned short sP[16][LPAD];

  int qrow = qt * 16 + nl;
  if (qrow > Nv - 1) qrow = Nv - 1;      // clamp; dead rows never written back
  const unsigned short* qbase =
      qkv + ((long long)(b * Nv + qrow) * (3 * Cv) + h * Dh);
  Frag a0, a1;
  load_a(a0, qbase, half);
  load_a(a1, qbase + 32, half);

  // --- S = (Q K^T) * Dh^-0.5 ---
  const unsigned short* kbase = Kg + (long long)bh * LPAD * Dh;
  for (int kt = 0; kt < LPAD / 16; kt++) {
    Frag b0, b1;
    const unsigned short* krow = kbase + (long long)(kt * 16 + nl) * Dh;
    load_b(b0, krow, half);
    load_b(b1, krow + 32, half);
    v8f acc = {};
    acc = wmma_bf16(a0, b0, acc);
    acc = wmma_bf16(a1, b1, acc);
#pragma unroll
    for (int r = 0; r < 8; r++)
      sS[r + 8 * half][kt * 16 + nl] = acc[r] * 0.125f;   // 1/sqrt(64)
  }
  __syncthreads();

  // --- row softmax over first KEEP cols, zero the pad ---
  for (int row = 0; row < 16; row++) {
    float mx = -1e30f;
    for (int c = lane; c < KEEP; c += 32) mx = fmaxf(mx, sS[row][c]);
    for (int o = 16; o > 0; o >>= 1) mx = fmaxf(mx, __shfl_xor(mx, o, 32));
    float sum = 0.f;
    for (int c = lane; c < LPAD; c += 32) {
      float e = (c < KEEP) ? __expf(sS[row][c] - mx) : 0.f;
      sS[row][c] = e;
      sum += e;
    }
    for (int o = 16; o > 0; o >>= 1) sum += __shfl_xor(sum, o, 32);
    float inv = 1.f / sum;
    for (int c = lane; c < LPAD; c += 32) sP[row][c] = f2bf(sS[row][c] * inv);
  }
  __syncthreads();

  // --- O = P @ V  (A fragments from LDS, B from VgT) ---
  v8f o0 = {}, o1 = {}, o2 = {}, o3 = {};
  const unsigned short* vbase = VgT + (long long)bh * Dh * LPAD;
  for (int c = 0; c < LPAD / 32; c++) {
    Frag pa;
    load_a(pa, &sP[nl][c * 32], half);
#pragma unroll
    for (int nt = 0; nt < 4; nt++) {
      Frag vb;
      load_b(vb, vbase + (long long)(nt * 16 + nl) * LPAD + c * 32, half);
      v8f& o = (nt == 0 ? o0 : nt == 1 ? o1 : nt == 2 ? o2 : o3);
      o = wmma_bf16(pa, vb, o);
    }
  }
#pragma unroll
  for (int r = 0; r < 8; r++) {
    const int row = qt * 16 + r + 8 * half;
    if (row < Nv) {
      long long ob = (long long)(b * Nv + row) * Cv + h * Dh;
      aout[ob + 0  + nl] = f2bf(o0[r]);
      aout[ob + 16 + nl] = f2bf(o1[r]);
      aout[ob + 32 + nl] = f2bf(o2[r]);
      aout[ob + 48 + nl] = f2bf(o3[r]);
    }
  }
}

// ---------------------------------------------------------------------------
extern "C" void kernel_launch(void* const* d_in, const int* in_sizes, int n_in,
                              void* d_out, int out_size, void* d_ws,
                              size_t ws_size, hipStream_t stream) {
  const float* x      = (const float*)d_in[0];
  const float* ln1_g  = (const float*)d_in[1];
  const float* ln1_b  = (const float*)d_in[2];
  const float* w_qkv  = (const float*)d_in[3];
  const float* w_proj = (const float*)d_in[4];
  const float* b_proj = (const float*)d_in[5];
  const float* ln2_g  = (const float*)d_in[6];
  const float* ln2_b  = (const float*)d_in[7];
  const float* w_fc1  = (const float*)d_in[8];
  const float* b_fc1  = (const float*)d_in[9];
  const float* w_fc2  = (const float*)d_in[10];
  const float* b_fc2  = (const float*)d_in[11];
  float* out = (float*)d_out;

  char* ws = (char*)d_ws;
  size_t off = 0;
  auto alloc = [&](size_t bytes) {
    off = (off + 255) & ~(size_t)255;
    void* p = ws + off;
    off += bytes;
    return p;
  };
  unsigned short* wqkvT  = (unsigned short*)alloc((size_t)3 * Cv * Cv * 2);
  unsigned short* wprojT = (unsigned short*)alloc((size_t)Cv * Cv * 2);
  unsigned short* wfc1T  = (unsigned short*)alloc((size_t)Hdim * Cv * 2);
  unsigned short* wfc2T  = (unsigned short*)alloc((size_t)Cv * Hdim * 2);
  unsigned short* hbf    = (unsigned short*)alloc((size_t)Mrows * Cv * 2);
  unsigned short* qkvbf  = (unsigned short*)alloc((size_t)Mrows * 3 * Cv * 2);
  float* scores          = (float*)alloc((size_t)Bv * Hh * 784 * 4);
  int* kidx              = (int*)alloc((size_t)Bv * Hh * SEL * 4);
  unsigned short* Kg     = (unsigned short*)alloc((size_t)Bv * Hh * LPAD * Dh * 2);
  unsigned short* VgT    = (unsigned short*)alloc((size_t)Bv * Hh * LPAD * Dh * 2);
  unsigned short* abf    = (unsigned short*)alloc((size_t)Mrows * Cv * 2);
  float* x2              = (float*)alloc((size_t)Mrows * Cv * 4);
  unsigned short* h2bf   = (unsigned short*)alloc((size_t)Mrows * Cv * 2);
  unsigned short* m1bf   = (unsigned short*)alloc((size_t)Mrows * Hdim * 2);

  auto cvt = [&](const float* W, unsigned short* Wt, int K, int N) {
    long long total = (long long)K * N;
    cvt_transpose<<<(unsigned)((total + 255) / 256), 256, 0, stream>>>(W, Wt, K, N);
  };
  cvt(w_qkv, wqkvT, Cv, 3 * Cv);
  cvt(w_proj, wprojT, Cv, Cv);
  cvt(w_fc1, wfc1T, Cv, Hdim);
  cvt(w_fc2, wfc2T, Hdim, Cv);

  const int MG = (Mrows + 63) / 64;   // 197 macro M tiles (last clamped)
  // LN1
  layernorm_bf16<<<Mrows, 256, 0, stream>>>(x, ln1_g, ln1_b, hbf);
  // QKV = h @ Wqkv  -> bf16
  gemm_wmma<0><<<dim3(MG, (3 * Cv) / 128), 256, 0, stream>>>(
      hbf, wqkvT, nullptr, nullptr, qkvbf, nullptr, Cv, 3 * Cv);
  // Token pruning
  knorm_scores<<<(Bv * Hh * 784 + 255) / 256, 256, 0, stream>>>(qkvbf, scores);
  select_topk<<<Bv * Hh, 256, 0, stream>>>(scores, kidx);
  gather_kv<<<dim3(LPAD, Bv * Hh), Dh, 0, stream>>>(qkvbf, kidx, Kg, VgT);
  // Fused attention
  attention_kernel<<<dim3((Nv + 15) / 16, Hh, Bv), 32, 0, stream>>>(qkvbf, Kg,
                                                                    VgT, abf);
  // x2 = x + attn @ Wproj + b_proj  (f32)
  gemm_wmma<2><<<dim3(MG, Cv / 128), 256, 0, stream>>>(
      abf, wprojT, b_proj, x, nullptr, x2, Cv, Cv);
  // LN2
  layernorm_bf16<<<Mrows, 256, 0, stream>>>(x2, ln2_g, ln2_b, h2bf);
  // m1 = gelu(h2 @ Wfc1 + b_fc1) -> bf16
  gemm_wmma<1><<<dim3(MG, Hdim / 128), 256, 0, stream>>>(
      h2bf, wfc1T, b_fc1, nullptr, m1bf, nullptr, Cv, Hdim);
  // out = x2 + m1 @ Wfc2 + b_fc2  (f32)
  gemm_wmma<2><<<dim3(MG, Cv / 128), 256, 0, stream>>>(
      m1bf, wfc2T, b_fc2, x2, nullptr, out, Hdim, Cv);
}
